// AFNO2D_70609262346721
// MI455X (gfx1250) — matrix operand order repaired and checked by reading, not compiled, not run on hardware
//
#include <hip/hip_runtime.h>
#include <hip/hip_bf16.h>

// ---------------------------------------------------------------------------
// AFNO2D on gfx1250: all FFT stages recast as dense DFT GEMMs so every stage
// runs on v_wmma_f32_16x16x32_f16. Intermediates stored as f16 planes.
// B operands staged in LDS in WMMA-fragment order -> ds_load_b128 pairs.
// ---------------------------------------------------------------------------

typedef __attribute__((ext_vector_type(16))) _Float16 v16h;
typedef __attribute__((ext_vector_type(8)))  float    v8f;

#define B_    4
#define H_    128
#define W_    128
#define C_    768
#define WF_   65
#define NB_   8
#define BLK_  96
#define LDP_  104          // padded LDS leading dim for row-major A tiles
#define NT_   192          // 6 wave32 per workgroup, one 16-col tile per wave

// ---------------- WMMA helpers (A/B: contiguous k = 16h+s per lane half; D per CDNA5 ISA 7.12.2) --------------

static __device__ __forceinline__ v8f wmma16(v16h a, v16h b, v8f c) {
  // D = A(16x32 f16) * B(32x16 f16) + C(16x16 f32)
  return __builtin_amdgcn_wmma_f32_16x16x32_f16(false, a, false, b,
                                                (short)0, c, false, false);
}

// A fragment: A[m0+.., k0+..] (16x32 tile, row-major source, leading dim ld).
// Per lane this is two contiguous 8-half runs -> compiler emits 2x b128 loads.
static __device__ __forceinline__ v16h frag_a(const _Float16* p, int ld,
                                              int m0, int k0) {
  const int lane = threadIdx.x & 31;
  const int m    = m0 + (lane & 15);
  const int kb   = (lane & 16) ? 16 : 0;     // lanes 16-31 hold K 16-31
  v16h f;
#pragma unroll
  for (int e = 0; e < 16; ++e) {
    const int k = k0 + kb + e;
    f[e] = p[m * ld + k];
  }
  return f;
}

// ---- packed B-fragment buffers: layout [ntile][kchunk][lane][16 halves] ----
// Each lane's fragment is one aligned contiguous 32B slot (2x ds_load_b128).

#define FB_IDX(nt, kt, lane, NKT) ((((nt) * (NKT) + (kt)) * 32 + (lane)) * 16)

static __device__ __forceinline__ void fb_store(_Float16* fb, int NKT,
                                                int k, int c, _Float16 v) {
  const int nt   = c >> 4, n = c & 15;
  const int kt   = k >> 5;
  const int lane = ((k >> 4) & 1) * 16 + n;  // B frag: lanes 16-31 hold K 16-31
  fb[FB_IDX(nt, kt, lane, NKT) + (k & 15)] = v;
}

static __device__ __forceinline__ v16h fb_load(const _Float16* fb, int NKT,
                                               int nt, int kt) {
  const int lane = threadIdx.x & 31;
  return *(const v16h*)(fb + FB_IDX(nt, kt, lane, NKT));
}

// D layout: lane = (m/8)*16 + n ; VGPR r holds row m = (lane>>4)*8 + r
#define D_ROWCOL() \
  const int lane = threadIdx.x & 31; \
  const int dn = lane & 15; \
  const int dm = (lane & 16) ? 8 : 0;

// ---------------- Table init: DFT / iDFT / irfft-synthesis matrices ---------

__global__ void init_tables(_Float16* Wr, _Float16* Wi,
                            _Float16* Hr, _Float16* Hi,
                            _Float16* IHr, _Float16* IHi,
                            _Float16* Gc, _Float16* Gs) {
  const float s   = 0.08838834764831845f;          // 1/sqrt(128)  (ortho)
  const float stp = 6.283185307179586f / 128.0f;   // 2*pi/N
  const int tid = blockIdx.x * blockDim.x + threadIdx.x;
  const int nth = gridDim.x * blockDim.x;

  // rfft along W: 80x128 (rows 65..79 zero-padded)
  for (int idx = tid; idx < 80 * 128; idx += nth) {
    const int k = idx / 128, w = idx % 128;
    if (k < WF_) {
      const float a = (float)((k * w) & 127) * stp;
      Wr[idx] = (_Float16)( __cosf(a) * s);
      Wi[idx] = (_Float16)(-__sinf(a) * s);
    } else { Wr[idx] = (_Float16)0.f; Wi[idx] = (_Float16)0.f; }
  }
  // fft / ifft along H: 128x128
  for (int idx = tid; idx < 128 * 128; idx += nth) {
    const int k = idx / 128, h = idx % 128;
    const float a = (float)((k * h) & 127) * stp;
    const float c = __cosf(a) * s, sn = __sinf(a) * s;
    Hr[idx]  = (_Float16)c;  Hi[idx]  = (_Float16)(-sn);
    IHr[idx] = (_Float16)c;  IHi[idx] = (_Float16)( sn);
  }
  // irfft along W (Hermitian-folded real synthesis): 128 x 96 (cols 65..95 zero)
  for (int idx = tid; idx < 128 * 96; idx += nth) {
    const int w = idx / 96, kw = idx % 96;
    if (kw < WF_) {
      const float amp = (kw == 0 || kw == 64) ? 1.0f : 2.0f;
      const float a = (float)((w * kw) & 127) * stp;
      Gc[idx] = (_Float16)( amp * __cosf(a) * s);
      Gs[idx] = (_Float16)(-amp * __sinf(a) * s);
    } else { Gc[idx] = (_Float16)0.f; Gs[idx] = (_Float16)0.f; }
  }
}

// ---------------- Stage A: rfft along W  (real GEMM, M=65pad80,K=128,N=768) --
// grid (B*H, 8 column panels of 96), 192 threads

__global__ __launch_bounds__(NT_) void
stage_rfftw(const float* __restrict__ x,
            const _Float16* __restrict__ Wr, const _Float16* __restrict__ Wi,
            _Float16* __restrict__ Ar, _Float16* __restrict__ Ai) {
  __shared__ __align__(32) _Float16 FB[6 * 4 * 512];   // packed B frags, K=128
  const int slice = blockIdx.x;                 // b*H + h
  const int cpan  = blockIdx.y * BLK_;
  const int tid   = threadIdx.x;

  for (int idx = tid; idx < W_ * BLK_; idx += NT_) {
    const int w = idx / BLK_, c = idx % BLK_;
    fb_store(FB, 4, w, c, (_Float16)x[(slice * W_ + w) * C_ + cpan + c]);
  }
  __syncthreads();

  const int nt = tid >> 5;                      // one 16-col tile per wave
  D_ROWCOL();
#pragma unroll 1
  for (int mt = 0; mt < 5; ++mt) {              // 80 rows of frequencies
    v8f accr = {}, acci = {};
#pragma unroll
    for (int kk = 0; kk < 4; ++kk) {
      const v16h bf = fb_load(FB, 4, nt, kk);
      accr = wmma16(frag_a(Wr, 128, mt * 16, kk * 32), bf, accr);
      acci = wmma16(frag_a(Wi, 128, mt * 16, kk * 32), bf, acci);
    }
#pragma unroll
    for (int r = 0; r < 8; ++r) {
      const int m = mt * 16 + dm + r;
      if (m < WF_) {
        const int o = (slice * WF_ + m) * C_ + cpan + nt * 16 + dn;
        Ar[o] = (_Float16)accr[r];
        Ai[o] = (_Float16)acci[r];
      }
    }
  }
}

// ---------------- Stage B/C: (i)fft along H (complex GEMM, M=K=128, N=768) --
// grid (B*65, 8 panels). Output rows indexed [b, kh, kw, c].

__global__ __launch_bounds__(NT_) void
stage_ffth(const _Float16* __restrict__ Sr, const _Float16* __restrict__ Si,
           const _Float16* __restrict__ Tr, const _Float16* __restrict__ Ti,
           _Float16* __restrict__ Dr, _Float16* __restrict__ Di) {
  __shared__ __align__(32) _Float16 FBr[6 * 4 * 512];
  __shared__ __align__(32) _Float16 FBi[6 * 4 * 512];
  const int b  = blockIdx.x / WF_;
  const int kw = blockIdx.x % WF_;
  const int cpan = blockIdx.y * BLK_;
  const int tid  = threadIdx.x;
  const int base = (b * H_ * WF_ + kw) * C_ + cpan;   // + h*WF_*C_ per row

  for (int idx = tid; idx < H_ * BLK_; idx += NT_) {
    const int h = idx / BLK_, c = idx % BLK_;
    const int g = base + h * (WF_ * C_) + c;
    fb_store(FBr, 4, h, c, Sr[g]);
    fb_store(FBi, 4, h, c, Si[g]);
  }
  __syncthreads();

  const int nt = tid >> 5;
  D_ROWCOL();
#pragma unroll 1
  for (int mt = 0; mt < 8; ++mt) {
    v8f aRR = {}, aII = {}, aRI = {}, aIR = {};
#pragma unroll
    for (int kk = 0; kk < 4; ++kk) {
      const v16h ar = frag_a(Tr, 128, mt * 16, kk * 32);
      const v16h ai = frag_a(Ti, 128, mt * 16, kk * 32);
      const v16h br = fb_load(FBr, 4, nt, kk);
      const v16h bi = fb_load(FBi, 4, nt, kk);
      aRR = wmma16(ar, br, aRR);  aII = wmma16(ai, bi, aII);
      aRI = wmma16(ar, bi, aRI);  aIR = wmma16(ai, br, aIR);
    }
#pragma unroll
    for (int r = 0; r < 8; ++r) {
      const int m = mt * 16 + dm + r;                 // kh
      const int o = (b * H_ * WF_ + m * WF_ + kw) * C_ + cpan + nt * 16 + dn;
      Dr[o] = (_Float16)(aRR[r] - aII[r]);
      Di[o] = (_Float16)(aRI[r] + aIR[r]);
    }
  }
}

// ---------------- Fused block-diagonal complex MLP ---------------------------
// grid (2080 row-tiles of 16, 8 blocks). Layer1: ReLU, Layer2: softshrink.

static __device__ __forceinline__ float softshrinkf(float v) {
  const float l = 0.01f;
  return v > l ? v - l : (v < -l ? v + l : 0.0f);
}

__global__ __launch_bounds__(NT_) void
mlp_block(const _Float16* __restrict__ Yr, const _Float16* __restrict__ Yi,
          const float* __restrict__ w1, const float* __restrict__ b1,
          const float* __restrict__ w2, const float* __restrict__ b2,
          _Float16* __restrict__ Zr, _Float16* __restrict__ Zi) {
  __shared__ __align__(32) _Float16 Wfr[6 * 3 * 512];  // packed weight frags (re)
  __shared__ __align__(32) _Float16 Wfi[6 * 3 * 512];  // packed weight frags (im)
  __shared__ _Float16 Xrs[16][LDP_];
  __shared__ _Float16 Xis[16][LDP_];
  __shared__ _Float16 Ors[16][LDP_];
  __shared__ _Float16 Ois[16][LDP_];

  const int mtile = blockIdx.x;
  const int blk   = blockIdx.y;
  const int tid   = threadIdx.x;
  const int cbase = blk * BLK_;

  // stage X tile (16 rows x 96) row-major + layer-1 weights fragment-packed
  for (int idx = tid; idx < 16 * BLK_; idx += NT_) {
    const int m = idx / BLK_, i = idx % BLK_;
    const int g = (mtile * 16 + m) * C_ + cbase + i;
    Xrs[m][i] = Yr[g];
    Xis[m][i] = Yi[g];
  }
  const int w1r = (0 * NB_ + blk) * BLK_ * BLK_;
  const int w1i = (1 * NB_ + blk) * BLK_ * BLK_;
  for (int idx = tid; idx < BLK_ * BLK_; idx += NT_) {
    const int k = idx / BLK_, n = idx % BLK_;          // w1 layout [i][o]
    fb_store(Wfr, 3, k, n, (_Float16)w1[w1r + idx]);
    fb_store(Wfi, 3, k, n, (_Float16)w1[w1i + idx]);
  }
  // warm layer-2 weights into cache while layer 1 runs
  const int w2r = (0 * NB_ + blk) * BLK_ * BLK_;
  const int w2i = (1 * NB_ + blk) * BLK_ * BLK_;
  for (int q = tid * 16; q < BLK_ * BLK_; q += NT_ * 16) {
    __builtin_prefetch(&w2[w2r + q], 0, 3);
    __builtin_prefetch(&w2[w2i + q], 0, 3);
  }
  __syncthreads();

  const int nt = tid >> 5;
  D_ROWCOL();
  const int o = nt * 16 + dn;                     // output column (0..95)

  // ---- layer 1:  o1 = relu( X * W1 + b1 )  (complex)
  {
    v8f aRR = {}, aII = {}, aIR = {}, aRI = {};
#pragma unroll
    for (int kk = 0; kk < 3; ++kk) {
      const v16h ar = frag_a(&Xrs[0][0], LDP_, 0, kk * 32);
      const v16h ai = frag_a(&Xis[0][0], LDP_, 0, kk * 32);
      const v16h br = fb_load(Wfr, 3, nt, kk);
      const v16h bi = fb_load(Wfi, 3, nt, kk);
      aRR = wmma16(ar, br, aRR);  aII = wmma16(ai, bi, aII);
      aIR = wmma16(ai, br, aIR);  aRI = wmma16(ar, bi, aRI);
    }
    const float br_ = b1[(0 * NB_ + blk) * BLK_ + o];
    const float bi_ = b1[(1 * NB_ + blk) * BLK_ + o];
#pragma unroll
    for (int r = 0; r < 8; ++r) {
      Ors[dm + r][o] = (_Float16)fmaxf(aRR[r] - aII[r] + br_, 0.0f);
      Ois[dm + r][o] = (_Float16)fmaxf(aIR[r] + aRI[r] + bi_, 0.0f);
    }
  }
  __syncthreads();
  // overwrite weight LDS with layer-2 weights ([o][i] -> B frag [k=o][n=i])
  for (int idx = tid; idx < BLK_ * BLK_; idx += NT_) {
    const int k = idx / BLK_, n = idx % BLK_;
    fb_store(Wfr, 3, k, n, (_Float16)w2[w2r + idx]);
    fb_store(Wfi, 3, k, n, (_Float16)w2[w2i + idx]);
  }
  __syncthreads();

  // ---- layer 2:  z = softshrink( o1 * W2 + b2 )  (complex)
  {
    v8f aRR = {}, aII = {}, aIR = {}, aRI = {};
#pragma unroll
    for (int kk = 0; kk < 3; ++kk) {
      const v16h ar = frag_a(&Ors[0][0], LDP_, 0, kk * 32);
      const v16h ai = frag_a(&Ois[0][0], LDP_, 0, kk * 32);
      const v16h br = fb_load(Wfr, 3, nt, kk);
      const v16h bi = fb_load(Wfi, 3, nt, kk);
      aRR = wmma16(ar, br, aRR);  aII = wmma16(ai, bi, aII);
      aIR = wmma16(ai, br, aIR);  aRI = wmma16(ar, bi, aRI);
    }
    const float br_ = b2[(0 * NB_ + blk) * BLK_ + o];
    const float bi_ = b2[(1 * NB_ + blk) * BLK_ + o];
#pragma unroll
    for (int r = 0; r < 8; ++r) {
      const int g = (mtile * 16 + dm + r) * C_ + cbase + o;
      Zr[g] = (_Float16)softshrinkf(aRR[r] - aII[r] + br_);
      Zi[g] = (_Float16)softshrinkf(aIR[r] + aRI[r] + bi_);
    }
  }
}

// ---------------- Stage D: irfft along W + residual (M=128,K=65pad96,N=768) -
// grid (B*H, 8 panels)

__global__ __launch_bounds__(NT_) void
stage_irfftw(const _Float16* __restrict__ Ur, const _Float16* __restrict__ Ui,
             const _Float16* __restrict__ Gc, const _Float16* __restrict__ Gs,
             const float* __restrict__ x, float* __restrict__ out) {
  __shared__ __align__(32) _Float16 FBr[6 * 3 * 512];  // kw rows padded to 96
  __shared__ __align__(32) _Float16 FBi[6 * 3 * 512];
  const int slice = blockIdx.x;          // b*H + h
  const int cpan  = blockIdx.y * BLK_;
  const int tid   = threadIdx.x;

  for (int idx = tid; idx < BLK_ * BLK_; idx += NT_) {
    const int kw = idx / BLK_, c = idx % BLK_;
    _Float16 vr = (_Float16)0.f, vi = (_Float16)0.f;
    if (kw < WF_) {
      const int g = (slice * WF_ + kw) * C_ + cpan + c;
      vr = Ur[g];
      vi = Ui[g];
    }
    fb_store(FBr, 3, kw, c, vr);
    fb_store(FBi, 3, kw, c, vi);
  }
  __syncthreads();

  const int nt = tid >> 5;
  D_ROWCOL();
#pragma unroll 1
  for (int mt = 0; mt < 8; ++mt) {       // 128 output w positions
    v8f acc = {};
#pragma unroll
    for (int kk = 0; kk < 3; ++kk) {
      acc = wmma16(frag_a(Gc, 96, mt * 16, kk * 32), fb_load(FBr, 3, nt, kk), acc);
      acc = wmma16(frag_a(Gs, 96, mt * 16, kk * 32), fb_load(FBi, 3, nt, kk), acc);
    }
#pragma unroll
    for (int r = 0; r < 8; ++r) {
      const int w = mt * 16 + dm + r;
      const int g = (slice * W_ + w) * C_ + cpan + nt * 16 + dn;
      out[g] = acc[r] + x[g];            // residual
    }
  }
}

// ---------------------------------------------------------------------------

extern "C" void kernel_launch(void* const* d_in, const int* in_sizes, int n_in,
                              void* d_out, int out_size, void* d_ws, size_t ws_size,
                              hipStream_t stream) {
  const float* x  = (const float*)d_in[0];
  const float* w1 = (const float*)d_in[1];
  const float* b1 = (const float*)d_in[2];
  const float* w2 = (const float*)d_in[3];
  const float* b2 = (const float*)d_in[4];
  float* out = (float*)d_out;

  char* ws = (char*)d_ws;
  size_t off = 0;
  auto alloc = [&](size_t elems) -> _Float16* {
    _Float16* p = (_Float16*)(ws + off);
    off += (elems * sizeof(_Float16) + 255) & ~(size_t)255;
    return p;
  };
  // DFT tables
  _Float16* tWr  = alloc(80 * 128);
  _Float16* tWi  = alloc(80 * 128);
  _Float16* tHr  = alloc(128 * 128);
  _Float16* tHi  = alloc(128 * 128);
  _Float16* tIHr = alloc(128 * 128);
  _Float16* tIHi = alloc(128 * 128);
  _Float16* tGc  = alloc(128 * 96);
  _Float16* tGs  = alloc(128 * 96);
  // f16 intermediate planes [B,128,65,768]; two ping-pong complex pairs
  const size_t NE = (size_t)B_ * H_ * WF_ * C_;
  _Float16* P0r = alloc(NE);
  _Float16* P0i = alloc(NE);
  _Float16* P1r = alloc(NE);
  _Float16* P1i = alloc(NE);

  init_tables<<<dim3(64), 256, 0, stream>>>(tWr, tWi, tHr, tHi, tIHr, tIHi, tGc, tGs);

  // rfft2:  W axis, then H axis
  stage_rfftw<<<dim3(B_ * H_, 8), NT_, 0, stream>>>(x, tWr, tWi, P0r, P0i);
  stage_ffth <<<dim3(B_ * WF_, 8), NT_, 0, stream>>>(P0r, P0i, tHr, tHi, P1r, P1i);

  // block-diagonal complex MLP over 33280 position rows
  mlp_block  <<<dim3((B_ * H_ * WF_) / 16, NB_), NT_, 0, stream>>>(
      P1r, P1i, w1, b1, w2, b2, P0r, P0i);

  // irfft2:  H axis inverse, then W-axis real synthesis + residual
  stage_ffth <<<dim3(B_ * WF_, 8), NT_, 0, stream>>>(P0r, P0i, tIHr, tIHi, P1r, P1i);
  stage_irfftw<<<dim3(B_ * H_, 8), NT_, 0, stream>>>(P1r, P1i, tGc, tGs, x, out);
}
